// MultiHeadAttention_5806795784479
// MI455X (gfx1250) — matrix operand order repaired and checked by reading, not compiled, not run on hardware
//
#include <hip/hip_runtime.h>
#include <hip/hip_bf16.h>

#define EMBED 1024
#define HEADS 16
#define HDIM  64
#define OUTD  1024
#define LSEQ  1024
#define BATCH 4

typedef __attribute__((ext_vector_type(16))) _Float16 v16h;
typedef __attribute__((ext_vector_type(2)))  _Float16 h2;
typedef __attribute__((ext_vector_type(8)))  float    v8f;

union Frag {
    uint4 u[2];
    v16h  h;
};

static __device__ __forceinline__ h2 pack2(float a, float b) {
#if __has_builtin(__builtin_amdgcn_cvt_pkrtz)
    return __builtin_bit_cast(h2, __builtin_amdgcn_cvt_pkrtz(a, b));
#else
    h2 r; r.x = (_Float16)a; r.y = (_Float16)b; return r;
#endif
}

// exp(tanh(x)) via hardware exp2/rcp:
//   tanh(x) = 1 - 2*rcp(exp2(2x*log2e) + 1) ; exp(t) = exp2(t*log2e)
// Saturates correctly for |x| large (exp2->0/inf => tanh->-1/+1).
static __device__ __forceinline__ float exp_tanh(float x) {
    const float L2E = 1.4426950408889634f;
    float u = __builtin_amdgcn_exp2f(x * (2.f * L2E));
    float t = 1.f - 2.f * __builtin_amdgcn_rcpf(u + 1.f);
    return __builtin_amdgcn_exp2f(t * L2E);
}

// ---------------------------------------------------------------------------
// Kernel 1: fold attn_w into Wq/Wk:  Wqp[e][n] = sum_d Wq[e, n*64+d]*w1[d]
// ---------------------------------------------------------------------------
__global__ __launch_bounds__(256) void prep_kernel(
    const float* __restrict__ Wq, const float* __restrict__ Wk,
    const float* __restrict__ bq, const float* __restrict__ attn_w,
    float* __restrict__ Wqp, float* __restrict__ Wkp, float* __restrict__ bqp)
{
    int gid = blockIdx.x * 256 + threadIdx.x;
    if (gid < EMBED * HEADS) {
        int e = gid >> 4, n = gid & 15;
        float a1 = 0.f, a2 = 0.f;
        const float* wq = Wq + (size_t)e * (HEADS * HDIM) + n * HDIM;
        const float* wk = Wk + (size_t)e * (HEADS * HDIM) + n * HDIM;
        for (int d = 0; d < HDIM; ++d) {
            a1 += wq[d] * attn_w[d];
            a2 += wk[d] * attn_w[HDIM + d];
        }
        Wqp[gid] = a1;
        Wkp[gid] = a2;
    }
    if (gid < HEADS) {
        float a = 0.f;
        for (int d = 0; d < HDIM; ++d) a += bq[gid * HDIM + d] * attn_w[d];
        bqp[gid] = a;
    }
}

// ---------------------------------------------------------------------------
// Kernel 2: sq = q @ Wqp + bqp ; sk = k @ Wkp    (skinny N=16 GEMMs)
// ---------------------------------------------------------------------------
__global__ __launch_bounds__(256) void sqsk_kernel(
    const float* __restrict__ q, const float* __restrict__ k,
    const float* __restrict__ Wqp, const float* __restrict__ Wkp,
    const float* __restrict__ bqp,
    float* __restrict__ sq, float* __restrict__ sk)
{
    int gid = blockIdx.x * 256 + threadIdx.x;    // (B*L*16) outputs
    int row = gid >> 4, n = gid & 15;
    const float* qr = q + (size_t)row * EMBED;
    const float* kr = k + (size_t)row * EMBED;
    float aq = 0.f, ak = 0.f;
    for (int e = 0; e < EMBED; ++e) {
        float wq = Wqp[e * 16 + n];
        float wk = Wkp[e * 16 + n];
        aq += qr[e] * wq;
        ak += kr[e] * wk;
    }
    sq[gid] = aq + bqp[n];
    sk[gid] = ak;
}

// ---------------------------------------------------------------------------
// WMMA GEMM: C = A(MxK fp32) * B(KxN fp32) [+bias]
//   MODE 0: Cf[m*N+n] = acc + bias[n]                      (fp32 out)
//   MODE 1: Ch in transposed f16 layout [b][gn][j] for vhT (f16 out)
// Block: 256 threads = 8 waves (2x4), tile 128x128, BK=32.
// ---------------------------------------------------------------------------
#define BM 128
#define BN 128
#define BK 32
#define LDT 40   // padded LDS row stride (f16 units), 80B keeps 16B alignment

template<int MODE>
__global__ __launch_bounds__(256) void gemm_wmma_kernel(
    const float* __restrict__ A, const float* __restrict__ B,
    const float* __restrict__ bias,
    float* __restrict__ Cf, _Float16* __restrict__ Ch,
    int M, int N, int K)
{
    __shared__ _Float16 As[BM][LDT];   // [m][k]
    __shared__ _Float16 Bs[BN][LDT];   // [n][k]  (transposed)

    int tid  = threadIdx.x;
    int lane = tid & 31, warp = tid >> 5;
    int wm = warp & 1;         // 0..1 -> 64-row slab
    int wn = warp >> 1;        // 0..3 -> 32-col slab
    int bm = blockIdx.y * BM, bn = blockIdx.x * BN;

    v8f acc[4][2] = {};

    int arow = tid >> 3;            // 0..31
    int ac4  = (tid & 7) * 4;       // 0..28
    int bk   = tid >> 5;            // 0..7
    int bc4  = (tid & 31) * 4;      // 0..124
    int kh   = (lane >> 4) * 8;     // A frag K-half select
    int kk2  = (lane >> 4) * 16;    // unused (B frags also use kh)
    int m16  = lane & 15;

    for (int k0 = 0; k0 < K; k0 += BK) {
        // ---- global fp32 -> LDS f16 : A tile 128x32 (64-bit packed stores)
        #pragma unroll
        for (int p = 0; p < 4; ++p) {
            int row = arow + p * 32;
            float4 f = *(const float4*)&A[(size_t)(bm + row) * K + k0 + ac4];
            union { h2 h[2]; uint2 u; } t;
            t.h[0] = pack2(f.x, f.y);
            t.h[1] = pack2(f.z, f.w);
            *(uint2*)&As[row][ac4] = t.u;
        }
        // ---- B tile 32x128, stored transposed [n][k]
        #pragma unroll
        for (int p = 0; p < 4; ++p) {
            int kk = bk + p * 8;
            float4 f = *(const float4*)&B[(size_t)(k0 + kk) * N + bn + bc4];
            Bs[bc4 + 0][kk] = (_Float16)f.x;
            Bs[bc4 + 1][kk] = (_Float16)f.y;
            Bs[bc4 + 2][kk] = (_Float16)f.z;
            Bs[bc4 + 3][kk] = (_Float16)f.w;
        }
        __syncthreads();

        Frag af[4], bf[2];
        #pragma unroll
        for (int mi = 0; mi < 4; ++mi) {
            int row = wm * 64 + mi * 16 + m16;
            af[mi].u[0] = *(const uint4*)&As[row][kh];       // K = kh..kh+7
            af[mi].u[1] = *(const uint4*)&As[row][kh + 16];  // K = kh+16..kh+23
        }
        #pragma unroll
        for (int ni = 0; ni < 2; ++ni) {
            int col = wn * 32 + ni * 16 + m16;
            bf[ni].u[0] = *(const uint4*)&Bs[col][kh];       // K = kh..kh+7
            bf[ni].u[1] = *(const uint4*)&Bs[col][kh + 16];  // K = kh+16..kh+23
        }
        #pragma unroll
        for (int mi = 0; mi < 4; ++mi)
            #pragma unroll
            for (int ni = 0; ni < 2; ++ni)
                acc[mi][ni] = __builtin_amdgcn_wmma_f32_16x16x32_f16(
                    false, af[mi].h, false, bf[ni].h,
                    (short)0, acc[mi][ni], false, false);
        __syncthreads();
    }

    // ---- epilogue (C layout: lane&15 = N, rows = (lane>>4)*8 + r)
    int mbase = (lane >> 4) * 8;
    #pragma unroll
    for (int mi = 0; mi < 4; ++mi) {
        #pragma unroll
        for (int ni = 0; ni < 2; ++ni) {
            int gm0 = bm + wm * 64 + mi * 16 + mbase;
            int gn  = bn + wn * 32 + ni * 16 + m16;
            if (MODE == 0) {
                float bb = bias[gn];
                #pragma unroll
                for (int r = 0; r < 8; ++r)
                    Cf[(size_t)(gm0 + r) * N + gn] = acc[mi][ni][r] + bb;
            } else {
                // vhT[b][gn][j], gm = b*1024 + j ; 8 consecutive j per lane
                int b = gm0 >> 10;
                int j = gm0 & 1023;
                union { h2 h[4]; uint4 u; } tmp;
                #pragma unroll
                for (int r = 0; r < 4; ++r)
                    tmp.h[r] = pack2(acc[mi][ni][2 * r], acc[mi][ni][2 * r + 1]);
                *(uint4*)&Ch[(size_t)b * (HEADS * HDIM * LSEQ)
                             + (size_t)gn * LSEQ + j] = tmp.u;
            }
        }
    }
}

// ---------------------------------------------------------------------------
// Kernel 4: attention. Per (b, head): O = P @ vh_n, P_ij = exp(tanh(sq_i+sk_j)),
// normalized by Z_i = sum_j P_ij (no max needed: tanh bounded).
// A-fragments built directly in registers; B from L2-resident vhT.
// ---------------------------------------------------------------------------
__global__ __launch_bounds__(256) void attn_kernel(
    const float* __restrict__ sq, const float* __restrict__ sk,
    const _Float16* __restrict__ vhT, float* __restrict__ ctx)
{
    __shared__ float sk_sh[LSEQ];
    int b = blockIdx.z, n = blockIdx.y, it = blockIdx.x;
    int tid = threadIdx.x, lane = tid & 31, warp = tid >> 5;

    for (int j = tid; j < LSEQ; j += 256)
        sk_sh[j] = sk[((size_t)b * LSEQ + j) * HEADS + n];
    __syncthreads();

    int m16 = lane & 15;
    int i   = it * 128 + warp * 16 + m16;        // this lane's query row
    float sqv = sq[((size_t)b * LSEQ + i) * HEADS + n];
    const _Float16* vbase = vhT + (size_t)b * (HEADS * HDIM * LSEQ)
                                + (size_t)n * HDIM * LSEQ;
    int kh  = (lane >> 4) * 8;
    int kk2 = (lane >> 4) * 16;

    v8f acc[4] = {};
    float z = 0.f;

    for (int j0 = 0; j0 < LSEQ; j0 += 32) {
        // issue all B-fragment loads first; VALU below hides their latency
        Frag bf[4];
        #pragma unroll
        for (int ni = 0; ni < 4; ++ni) {
            int d = ni * 16 + m16;
            const _Float16* src = vbase + (size_t)d * LSEQ + j0 + kh;
            bf[ni].u[0] = *(const uint4*)src;
            bf[ni].u[1] = *(const uint4*)(src + 16);
        }

        // A-frag: VGPR r pair holds K = kh + 2r (+8 for upper half), fully
        // unrolled constant-index packing (no scratch round-trips)
        union { h2 p[8]; v16h v; } af;
        #pragma unroll
        for (int r = 0; r < 8; ++r) {
            int jj = j0 + kh + 2 * r + ((r >= 4) ? 8 : 0);
            float p0 = exp_tanh(sqv + sk_sh[jj]);
            float p1 = exp_tanh(sqv + sk_sh[jj + 1]);
            z += p0 + p1;
            af.p[r] = pack2(p0, p1);
        }

        #pragma unroll
        for (int ni = 0; ni < 4; ++ni)
            acc[ni] = __builtin_amdgcn_wmma_f32_16x16x32_f16(
                false, af.v, false, bf[ni].h, (short)0, acc[ni], false, false);
    }

    // lane l and l^16 hold complementary K-halves of the same row M = lane&15
    z += __shfl_xor(z, 16, 32);
    // Z in [1024/e, 1024*e]: perfectly conditioned -> 1-ulp hardware rcp,
    // broadcast the reciprocal and multiply (no IEEE div expansion)
    float rz = __builtin_amdgcn_rcpf(z);

    int mbase = (lane >> 4) * 8;
    #pragma unroll
    for (int r = 0; r < 8; ++r) {
        int Mrow = mbase + r;
        float zr = __shfl(rz, Mrow, 32);  // lane Mrow holds 1/Z for row Mrow
        int gi = it * 128 + warp * 16 + Mrow;
        float* dst = ctx + ((size_t)b * LSEQ + gi) * (HEADS * HDIM) + n * HDIM;
        #pragma unroll
        for (int ni = 0; ni < 4; ++ni)
            dst[ni * 16 + m16] = acc[ni][r] * zr;
    }
}

// ---------------------------------------------------------------------------
// Kernel 6: LayerNorm over last dim (1024), in place on d_out
// ---------------------------------------------------------------------------
__global__ __launch_bounds__(256) void ln_kernel(
    float* __restrict__ out, const float* __restrict__ gamma,
    const float* __restrict__ beta)
{
    __shared__ float red[2][8];
    float* p = out + (size_t)blockIdx.x * OUTD;
    int tid = threadIdx.x, lane = tid & 31, warp = tid >> 5;

    float vals[4], s = 0.f, ss = 0.f;
    #pragma unroll
    for (int t = 0; t < 4; ++t) {
        float v = p[tid + t * 256];
        vals[t] = v; s += v; ss += v * v;
    }
    #pragma unroll
    for (int o = 16; o > 0; o >>= 1) {
        s += __shfl_xor(s, o, 32); ss += __shfl_xor(ss, o, 32);
    }
    if (lane == 0) { red[0][warp] = s; red[1][warp] = ss; }
    __syncthreads();
    if (warp == 0) {
        s  = (lane < 8) ? red[0][lane] : 0.f;
        ss = (lane < 8) ? red[1][lane] : 0.f;
        #pragma unroll
        for (int o = 4; o > 0; o >>= 1) {
            s += __shfl_xor(s, o, 32); ss += __shfl_xor(ss, o, 32);
        }
        if (lane == 0) { red[0][0] = s; red[1][0] = ss; }
    }
    __syncthreads();
    float mean = red[0][0] * (1.f / OUTD);
    float var  = red[1][0] * (1.f / OUTD) - mean * mean;
    float rstd = rsqrtf(var + 1e-6f);
    #pragma unroll
    for (int t = 0; t < 4; ++t) {
        int idx = tid + t * 256;
        p[idx] = (vals[t] - mean) * rstd * gamma[idx] + beta[idx];
    }
}

// ---------------------------------------------------------------------------
extern "C" void kernel_launch(void* const* d_in, const int* in_sizes, int n_in,
                              void* d_out, int out_size, void* d_ws, size_t ws_size,
                              hipStream_t stream) {
    const float* k_in  = (const float*)d_in[0];
    const float* q_in  = (const float*)d_in[1];
    const float* v_in  = (const float*)d_in[2];
    const float* Wq    = (const float*)d_in[3];
    const float* bq    = (const float*)d_in[4];
    const float* Wk    = (const float*)d_in[5];
    const float* Wv    = (const float*)d_in[6];
    const float* Wp    = (const float*)d_in[7];
    const float* bp    = (const float*)d_in[8];
    const float* aw    = (const float*)d_in[9];
    const float* gamma = (const float*)d_in[10];
    const float* beta  = (const float*)d_in[11];
    float* out = (float*)d_out;

    const int Mtot = BATCH * LSEQ;   // 4096

    // workspace carve-out (256B aligned), ~25 MB total
    char* ws = (char*)d_ws;
    size_t off = 0;
    auto carve = [&](size_t bytes) {
        void* p = ws + off;
        off = (off + bytes + 255) & ~(size_t)255;
        return p;
    };
    float*    Wqp = (float*)carve((size_t)EMBED * HEADS * 4);
    float*    Wkp = (float*)carve((size_t)EMBED * HEADS * 4);
    float*    bqp = (float*)carve(HEADS * 4);
    float*    sq  = (float*)carve((size_t)Mtot * HEADS * 4);
    float*    sk  = (float*)carve((size_t)Mtot * HEADS * 4);
    _Float16* vhT = (_Float16*)carve((size_t)BATCH * HEADS * HDIM * LSEQ * 2);
    float*    ctx = (float*)carve((size_t)Mtot * HEADS * HDIM * 4);

    // 1) fold attn_w into projection weights
    prep_kernel<<<(EMBED * HEADS + 255) / 256, 256, 0, stream>>>(
        Wq, Wk, bq, aw, Wqp, Wkp, bqp);

    // 2) per-head query/key scalars
    sqsk_kernel<<<(Mtot * HEADS) / 256, 256, 0, stream>>>(
        q_in, k_in, Wqp, Wkp, bqp, sq, sk);

    // 3) vh = v @ Wv  -> f16, transposed [b][head*64+d][j]
    gemm_wmma_kernel<1><<<dim3((HEADS * HDIM) / BN, Mtot / BM), 256, 0, stream>>>(
        v_in, Wv, nullptr, nullptr, vhT, Mtot, HEADS * HDIM, EMBED);

    // 4) additive attention with fused softmax (WMMA P@V)
    attn_kernel<<<dim3(LSEQ / 128, HEADS, BATCH), 256, 0, stream>>>(
        sq, sk, vhT, ctx);

    // 5) projection: out = ctx @ Wp + bp
    gemm_wmma_kernel<0><<<dim3(OUTD / BN, Mtot / BM), 256, 0, stream>>>(
        ctx, Wp, bp, out, nullptr, Mtot, OUTD, HEADS * HDIM);

    // 6) layernorm in place
    ln_kernel<<<Mtot, 256, 0, stream>>>(out, gamma, beta);
}
